// TransformerEncoderRPALayer_49821620633750
// MI455X (gfx1250) — matrix operand order repaired and checked by reading, not compiled, not run on hardware
//
#include <hip/hip_runtime.h>

// ---------------------------------------------------------------------------
// Problem constants (from reference)
// ---------------------------------------------------------------------------
constexpr int S_LEN   = 2048;
constexpr int BATCH_N = 2;
constexpr int D_MOD   = 512;
constexpr int N_HEAD  = 8;
constexpr int D_HEAD  = 64;
constexpr int D_FF    = 2048;
constexpr int MAX_LEN = 16384;

typedef __attribute__((ext_vector_type(16))) __bf16 v16bf;
typedef __attribute__((ext_vector_type(8)))  __bf16 v8bf;
typedef __attribute__((ext_vector_type(8)))  float  v8f;

// ---------------------------------------------------------------------------
// WMMA helpers
// ---------------------------------------------------------------------------
static __device__ inline v8f wmma_bf16(v16bf a, v16bf b, v8f c) {
#if defined(__gfx1250__)
  return __builtin_amdgcn_wmma_f32_16x16x32_bf16(false, a, false, b,
                                                 (short)0, c, false, false);
#else
  return c;  // host-compile stub
#endif
}

// Load a 16-element bf16 fragment as two contiguous 8-element (16B) chunks.
static __device__ inline v16bf load_run2(const __bf16* p0, const __bf16* p1) {
  v8bf lo = *(const v8bf*)p0;
  v8bf hi = *(const v8bf*)p1;
  v16bf r;
#pragma unroll
  for (int i = 0; i < 8; ++i) { r[i] = lo[i]; r[i + 8] = hi[i]; }
  return r;
}

// One wave computes a 16x64 f32 tile of A[16xK] * B^T (B stored transposed,
// i.e. Bt[n][k]), K in steps of 32 with 4 WMMAs per step.
// Software-pipelined: next k-step's fragments are issued before this step's
// WMMAs so s_wait_loadcnt waits are partial and VMEM overlaps the matrix pipe.
// A-frag per-lane layout: row = lane%16, K = e + 16*(lane/16)
// B-frag per-lane layout: col = lane%16, K = e + 16*(lane/16)
template <int K>
static __device__ inline void wmma_tile_16x64(const __bf16* A, int lda,
                                              const __bf16* Bt, int ldb,
                                              v8f acc[4], int lane) {
  const int mrow   = lane & 15;
  const int half8  = (lane >> 4) << 3;   // 0 or 8
  const int half16 = (lane >> 4) << 4;   // 0 or 16
  const __bf16* ar = A + (size_t)mrow * lda;

  v16bf a_cur = load_run2(ar + half16, ar + half16 + 8);
  v16bf b_cur[4];
#pragma unroll
  for (int f = 0; f < 4; ++f) {
    const __bf16* bp = Bt + (size_t)(f * 16 + mrow) * ldb + half16;
    b_cur[f] = load_run2(bp, bp + 8);
  }

  for (int k0 = 32; k0 < K; k0 += 32) {
    // prefetch next k-step
    const __bf16* ap = ar + k0;
    v16bf a_nxt = load_run2(ap + half16, ap + half16 + 8);
    v16bf b_nxt[4];
#pragma unroll
    for (int f = 0; f < 4; ++f) {
      const __bf16* bp = Bt + (size_t)(f * 16 + mrow) * ldb + k0 + half16;
      b_nxt[f] = load_run2(bp, bp + 8);
    }
    // consume current
#pragma unroll
    for (int f = 0; f < 4; ++f) acc[f] = wmma_bf16(a_cur, b_cur[f], acc[f]);
    a_cur = a_nxt;
#pragma unroll
    for (int f = 0; f < 4; ++f) b_cur[f] = b_nxt[f];
  }
#pragma unroll
  for (int f = 0; f < 4; ++f) acc[f] = wmma_bf16(a_cur, b_cur[f], acc[f]);
}

// ---------------------------------------------------------------------------
// Casting / packing kernels
// ---------------------------------------------------------------------------
// x[b][s][d] = src[s][b][d]  (batch-major activation matrix, bf16)
__global__ __launch_bounds__(256) void k_cast_x(const float* src, __bf16* xbf) {
  size_t idx = (size_t)blockIdx.x * 256 + threadIdx.x;
  int d = (int)(idx & (D_MOD - 1));
  int r = (int)(idx >> 9);
  int b = r / S_LEN, s = r % S_LEN;
  xbf[idx] = (__bf16)src[((size_t)s * BATCH_N + b) * D_MOD + d];
}

// dst[n][k] = (bf16) src[k][n]  (transposed bf16 weights)
__global__ __launch_bounds__(256) void k_tcast(const float* src, __bf16* dst,
                                               int K, int N) {
  size_t idx = (size_t)blockIdx.x * 256 + threadIdx.x;
  if (idx >= (size_t)K * N) return;
  int k = (int)(idx / N), n = (int)(idx % N);
  dst[(size_t)n * K + k] = (__bf16)src[idx];
}

// er[c][d] = Er[MAX_LEN - S + c][d] as bf16
__global__ __launch_bounds__(256) void k_cast_er(const float* Er, __bf16* erb) {
  size_t idx = (size_t)blockIdx.x * 256 + threadIdx.x;
  int d = (int)(idx & (D_HEAD - 1));
  int c = (int)(idx >> 6);
  erb[idx] = (__bf16)Er[((size_t)(MAX_LEN - S_LEN + c)) * D_HEAD + d];
}

// ---------------------------------------------------------------------------
// QKV projection (batch 0 only): [2048 x 512] @ [512 x 512] + bias
// z = 0:Q  1:K  2:V.  Q,K stored [h][s][dh]; V stored transposed [h][dh][s].
// ---------------------------------------------------------------------------
__global__ __launch_bounds__(32) void k_qkv(const __bf16* xbf,
                                            const __bf16* Wqt, const __bf16* Wkt,
                                            const __bf16* Wvt,
                                            const float* bq, const float* bk,
                                            const float* bv,
                                            __bf16* qb, __bf16* kb, __bf16* vT) {
  const int lane = threadIdx.x;
  const int c0 = blockIdx.x * 64;
  const int i0 = blockIdx.y * 16;
  const int z  = blockIdx.z;
  const __bf16* Wt  = (z == 0) ? Wqt : (z == 1) ? Wkt : Wvt;
  const float* bias = (z == 0) ? bq  : (z == 1) ? bk  : bv;

  v8f zero = {};
  v8f acc[4];
#pragma unroll
  for (int f = 0; f < 4; ++f) acc[f] = zero;

  wmma_tile_16x64<D_MOD>(xbf + (size_t)i0 * D_MOD, D_MOD,
                         Wt + (size_t)c0 * D_MOD, D_MOD, acc, lane);

  const int mrow = lane & 15;
  const int mhi  = (lane >> 4) << 3;
#pragma unroll
  for (int f = 0; f < 4; ++f) {
#pragma unroll
    for (int v = 0; v < 8; ++v) {
      int row = i0 + v + mhi;
      int c   = c0 + f * 16 + mrow;
      float val = acc[f][v] + bias[c];
      int hh = c >> 6, d = c & 63;
      if (z == 2)      vT[((size_t)hh * D_HEAD + d) * S_LEN + row] = (__bf16)val;
      else if (z == 1) kb[((size_t)hh * S_LEN + row) * D_HEAD + d] = (__bf16)val;
      else             qb[((size_t)hh * S_LEN + row) * D_HEAD + d] = (__bf16)val;
    }
  }
}

// ---------------------------------------------------------------------------
// QEr = Q @ Er^T, per head, materialized bf16 [h][s][c].
// 8*2048*2048*2B = 67 MB -> resident in MI455X's 192 MB L2 for the gathers.
// ---------------------------------------------------------------------------
__global__ __launch_bounds__(32) void k_qer(const __bf16* qb, const __bf16* erb,
                                            __bf16* qer) {
  const int lane = threadIdx.x;
  const int c0 = blockIdx.x * 64;
  const int i0 = blockIdx.y * 16;
  const int h  = blockIdx.z;

  v8f zero = {};
  v8f acc[4];
#pragma unroll
  for (int f = 0; f < 4; ++f) acc[f] = zero;

  wmma_tile_16x64<D_HEAD>(qb + ((size_t)h * S_LEN + i0) * D_HEAD, D_HEAD,
                          erb + (size_t)c0 * D_HEAD, D_HEAD, acc, lane);

  const int mrow = lane & 15;
  const int mhi  = (lane >> 4) << 3;
#pragma unroll
  for (int f = 0; f < 4; ++f) {
#pragma unroll
    for (int v = 0; v < 8; ++v) {
      int row = i0 + v + mhi;
      int c   = c0 + f * 16 + mrow;
      qer[((size_t)h * S_LEN + row) * S_LEN + c] = (__bf16)acc[f][v];
    }
  }
}

// Srel[i][j] reproducing the reference's skew (incl. wrap-around quirk):
//   t = j-i;  t<=0 -> QEr[i][S-1+t];  t==1 -> 0;  t>=2 -> QEr[i+1][t-2]
static __device__ inline float srel_val(const __bf16* qer_h, int i, int j) {
  int t = j - i;
  int row = (t <= 0) ? i : (i + 1);
  int col = (t <= 0) ? (S_LEN - 1 + t) : (t - 2);   // t==1 -> in-bounds aliased read
  float r = (float)qer_h[(size_t)row * S_LEN + col];
  return (t == 1) ? 0.f : r;
}

// ---------------------------------------------------------------------------
// Flash attention: one wave per (head, 16-row tile). Online softmax over
// 64 column tiles of 32. Scores + P.V via WMMA; Srel gathered from L2-hot QEr.
// V fragments are loaded at the top of each iteration so their latency is
// hidden under the softmax VALU section.
// ---------------------------------------------------------------------------
__global__ __launch_bounds__(32) void k_attn(const __bf16* qb, const __bf16* kb,
                                             const __bf16* vT, const __bf16* qer,
                                             float* sa) {
  const int lane = threadIdx.x;
  const int h  = blockIdx.y;
  const int i0 = blockIdx.x * 16;
  const __bf16* qer_h = qer + (size_t)h * S_LEN * S_LEN;

  __shared__ __align__(16) __bf16 pl[16 * 32];

  const int mrow   = lane & 15;
  const int half8  = (lane >> 4) << 3;
  const int half16 = (lane >> 4) << 4;
  const int mhi    = (lane >> 4) << 3;

  // Q A-fragments for K-steps d=[0,32) and [32,64)
  const __bf16* qrow = qb + ((size_t)h * S_LEN + i0 + mrow) * D_HEAD;
  v16bf aq0 = load_run2(qrow + half16, qrow + half16 + 8);
  v16bf aq1 = load_run2(qrow + 32 + half16, qrow + 40 + half16);

  v8f zero = {};
  float m[8], lsum[8];
  v8f co[4];
#pragma unroll
  for (int v = 0; v < 8; ++v) { m[v] = -1e30f; lsum[v] = 0.f; }
#pragma unroll
  for (int f = 0; f < 4; ++f) co[f] = zero;

  for (int j0 = 0; j0 < S_LEN; j0 += 32) {
    // ---- issue ALL VMEM for this iteration up front ----
    const __bf16* kr0 = kb + ((size_t)h * S_LEN + j0 + mrow) * D_HEAD + half16;
    v16bf b00 = load_run2(kr0, kr0 + 8);
    v16bf b01 = load_run2(kr0 + 32, kr0 + 40);
    const __bf16* kr1 = kb + ((size_t)h * S_LEN + j0 + 16 + mrow) * D_HEAD + half16;
    v16bf b10 = load_run2(kr1, kr1 + 8);
    v16bf b11 = load_run2(kr1 + 32, kr1 + 40);
    v16bf bv[4];
#pragma unroll
    for (int f = 0; f < 4; ++f) {
      const __bf16* vr = vT + ((size_t)h * D_HEAD + f * 16 + mrow) * S_LEN
                            + j0 + half16;
      bv[f] = load_run2(vr, vr + 8);   // consumed only by the P.V WMMAs below
    }

    // ---- S = Q K^T for a 16x32 score tile (two 16x16 C-frags) ----
    v8f s0 = zero, s1 = zero;
    s0 = wmma_bf16(aq0, b00, s0);
    s0 = wmma_bf16(aq1, b01, s0);
    s1 = wmma_bf16(aq0, b10, s1);
    s1 = wmma_bf16(aq1, b11, s1);

    // ---- + Srel, scale, online softmax ----
    float al[8];
#pragma unroll
    for (int v = 0; v < 8; ++v) {
      int M = v + mhi;
      int i = i0 + M;
      int ja = j0 + mrow;
      int jb = ja + 16;
      float x0 = (s0[v] + srel_val(qer_h, i, ja)) * 0.125f;
      float x1 = (s1[v] + srel_val(qer_h, i, jb)) * 0.125f;
      float t = fmaxf(x0, x1);
      t = fmaxf(t, __shfl_xor(t, 1));
      t = fmaxf(t, __shfl_xor(t, 2));
      t = fmaxf(t, __shfl_xor(t, 4));
      t = fmaxf(t, __shfl_xor(t, 8));
      float mn = fmaxf(m[v], t);
      float a  = __expf(m[v] - mn);
      m[v] = mn; al[v] = a;
      float p0 = __expf(x0 - mn);
      float p1 = __expf(x1 - mn);
      float rs = p0 + p1;
      rs += __shfl_xor(rs, 1);
      rs += __shfl_xor(rs, 2);
      rs += __shfl_xor(rs, 4);
      rs += __shfl_xor(rs, 8);
      lsum[v] = lsum[v] * a + rs;
      // C-layout -> LDS (row-major 16x32 bf16 P tile)
      pl[M * 32 + mrow]      = (__bf16)p0;
      pl[M * 32 + 16 + mrow] = (__bf16)p1;
    }
#pragma unroll
    for (int f = 0; f < 4; ++f)
#pragma unroll
      for (int v = 0; v < 8; ++v) co[f][v] *= al[v];

    __syncthreads();
    // P as A-fragment (16x32): O += P @ V   (V frags already in registers)
    const __bf16* pr = &pl[mrow * 32];
    v16bf ap = load_run2(pr + half16, pr + half16 + 8);
#pragma unroll
    for (int f = 0; f < 4; ++f) co[f] = wmma_bf16(ap, bv[f], co[f]);
    __syncthreads();
  }

  // ---- normalize + store sa[s][h*64+d] (f32) ----
#pragma unroll
  for (int f = 0; f < 4; ++f) {
#pragma unroll
    for (int v = 0; v < 8; ++v) {
      int row = i0 + v + mhi;
      int col = h * D_HEAD + f * 16 + mrow;
      sa[(size_t)row * D_MOD + col] = co[f][v] / lsum[v];
    }
  }
}

// ---------------------------------------------------------------------------
// LayerNorm 1: x1 = LN(x + sa_broadcast); also emits bf16 copy for the FFN.
// One wave per row; 16 elements/lane; full-wave shuffle reductions.
// ---------------------------------------------------------------------------
__global__ __launch_bounds__(32) void k_ln1(const float* src, const float* sa,
                                            const float* g, const float* be,
                                            float* x1, __bf16* x1bf) {
  const int r = blockIdx.x;
  const int b = r / S_LEN, s = r % S_LEN;
  const int lane = threadIdx.x;
  float xv[16];
  float sum = 0.f;
#pragma unroll
  for (int k = 0; k < 16; ++k) {
    int d = lane + 32 * k;
    float v = src[((size_t)s * BATCH_N + b) * D_MOD + d]
            + sa[(size_t)s * D_MOD + d];
    xv[k] = v; sum += v;
  }
  sum += __shfl_xor(sum, 1);  sum += __shfl_xor(sum, 2);
  sum += __shfl_xor(sum, 4);  sum += __shfl_xor(sum, 8);
  sum += __shfl_xor(sum, 16);
  float mu = sum * (1.f / D_MOD);
  float s2 = 0.f;
#pragma unroll
  for (int k = 0; k < 16; ++k) { float dv = xv[k] - mu; s2 += dv * dv; }
  s2 += __shfl_xor(s2, 1);  s2 += __shfl_xor(s2, 2);
  s2 += __shfl_xor(s2, 4);  s2 += __shfl_xor(s2, 8);
  s2 += __shfl_xor(s2, 16);
  float inv = rsqrtf(s2 * (1.f / D_MOD) + 1e-5f);
#pragma unroll
  for (int k = 0; k < 16; ++k) {
    int d = lane + 32 * k;
    float y = (xv[k] - mu) * inv * g[d] + be[d];
    x1[(size_t)r * D_MOD + d]   = y;
    x1bf[(size_t)r * D_MOD + d] = (__bf16)y;
  }
}

// ---------------------------------------------------------------------------
// FFN GEMMs
// ---------------------------------------------------------------------------
__global__ __launch_bounds__(32) void k_ff1(const __bf16* x1bf, const __bf16* W1t,
                                            const float* b1, __bf16* hbf) {
  const int lane = threadIdx.x;
  const int c0 = blockIdx.x * 64;
  const int r0 = blockIdx.y * 16;
  v8f zero = {};
  v8f acc[4];
#pragma unroll
  for (int f = 0; f < 4; ++f) acc[f] = zero;
  wmma_tile_16x64<D_MOD>(x1bf + (size_t)r0 * D_MOD, D_MOD,
                         W1t + (size_t)c0 * D_MOD, D_MOD, acc, lane);
  const int mrow = lane & 15;
  const int mhi  = (lane >> 4) << 3;
#pragma unroll
  for (int f = 0; f < 4; ++f) {
#pragma unroll
    for (int v = 0; v < 8; ++v) {
      int row = r0 + v + mhi;
      int c   = c0 + f * 16 + mrow;
      float val = fmaxf(acc[f][v] + b1[c], 0.f);   // ReLU
      hbf[(size_t)row * D_FF + c] = (__bf16)val;
    }
  }
}

__global__ __launch_bounds__(32) void k_ff2(const __bf16* hbf, const __bf16* W2t,
                                            const float* b2, float* ffw) {
  const int lane = threadIdx.x;
  const int c0 = blockIdx.x * 64;
  const int r0 = blockIdx.y * 16;
  v8f zero = {};
  v8f acc[4];
#pragma unroll
  for (int f = 0; f < 4; ++f) acc[f] = zero;
  wmma_tile_16x64<D_FF>(hbf + (size_t)r0 * D_FF, D_FF,
                        W2t + (size_t)c0 * D_FF, D_FF, acc, lane);
  const int mrow = lane & 15;
  const int mhi  = (lane >> 4) << 3;
#pragma unroll
  for (int f = 0; f < 4; ++f) {
#pragma unroll
    for (int v = 0; v < 8; ++v) {
      int row = r0 + v + mhi;
      int c   = c0 + f * 16 + mrow;
      ffw[(size_t)row * D_MOD + c] = acc[f][v] + b2[c];
    }
  }
}

// ---------------------------------------------------------------------------
// LayerNorm 2: out[s][b][d] = LN(x1 + ff)
// ---------------------------------------------------------------------------
__global__ __launch_bounds__(32) void k_ln2(const float* x1, const float* ffw,
                                            const float* g, const float* be,
                                            float* out) {
  const int r = blockIdx.x;
  const int b = r / S_LEN, s = r % S_LEN;
  const int lane = threadIdx.x;
  float xv[16];
  float sum = 0.f;
#pragma unroll
  for (int k = 0; k < 16; ++k) {
    int d = lane + 32 * k;
    float v = x1[(size_t)r * D_MOD + d] + ffw[(size_t)r * D_MOD + d];
    xv[k] = v; sum += v;
  }
  sum += __shfl_xor(sum, 1);  sum += __shfl_xor(sum, 2);
  sum += __shfl_xor(sum, 4);  sum += __shfl_xor(sum, 8);
  sum += __shfl_xor(sum, 16);
  float mu = sum * (1.f / D_MOD);
  float s2 = 0.f;
#pragma unroll
  for (int k = 0; k < 16; ++k) { float dv = xv[k] - mu; s2 += dv * dv; }
  s2 += __shfl_xor(s2, 1);  s2 += __shfl_xor(s2, 2);
  s2 += __shfl_xor(s2, 4);  s2 += __shfl_xor(s2, 8);
  s2 += __shfl_xor(s2, 16);
  float inv = rsqrtf(s2 * (1.f / D_MOD) + 1e-5f);
#pragma unroll
  for (int k = 0; k < 16; ++k) {
    int d = lane + 32 * k;
    float y = (xv[k] - mu) * inv * g[d] + be[d];
    out[((size_t)s * BATCH_N + b) * D_MOD + d] = y;
  }
}

// ---------------------------------------------------------------------------
// Host launcher
// ---------------------------------------------------------------------------
extern "C" void kernel_launch(void* const* d_in, const int* in_sizes, int n_in,
                              void* d_out, int out_size, void* d_ws, size_t ws_size,
                              hipStream_t stream) {
  const float* src = (const float*)d_in[0];
  const float* Wq  = (const float*)d_in[1];
  const float* bq  = (const float*)d_in[2];
  const float* Wk  = (const float*)d_in[3];
  const float* bk  = (const float*)d_in[4];
  const float* Wv  = (const float*)d_in[5];
  const float* bv  = (const float*)d_in[6];
  const float* Er  = (const float*)d_in[7];
  const float* W1  = (const float*)d_in[8];
  const float* b1  = (const float*)d_in[9];
  const float* W2  = (const float*)d_in[10];
  const float* b2  = (const float*)d_in[11];
  const float* g1  = (const float*)d_in[12];
  const float* be1 = (const float*)d_in[13];
  const float* g2  = (const float*)d_in[14];
  const float* be2 = (const float*)d_in[15];
  float* out = (float*)d_out;

  char* base = (char*)d_ws;
  size_t off = 0;
  auto alloc = [&](size_t bytes) -> char* {
    char* p = base + off;
    off += (bytes + 255) & ~(size_t)255;
    return p;
  };
  const size_t RS = (size_t)BATCH_N * S_LEN;  // 4096 rows

  __bf16* xbf  = (__bf16*)alloc(RS * D_MOD * 2);                      //  4 MB
  __bf16* Wqt  = (__bf16*)alloc((size_t)D_MOD * D_MOD * 2);
  __bf16* Wkt  = (__bf16*)alloc((size_t)D_MOD * D_MOD * 2);
  __bf16* Wvt  = (__bf16*)alloc((size_t)D_MOD * D_MOD * 2);
  __bf16* W1t  = (__bf16*)alloc((size_t)D_FF * D_MOD * 2);            //  2 MB
  __bf16* W2t  = (__bf16*)alloc((size_t)D_MOD * D_FF * 2);            //  2 MB
  __bf16* erb  = (__bf16*)alloc((size_t)S_LEN * D_HEAD * 2);
  __bf16* qb   = (__bf16*)alloc((size_t)N_HEAD * S_LEN * D_HEAD * 2); //  2 MB
  __bf16* kb   = (__bf16*)alloc((size_t)N_HEAD * S_LEN * D_HEAD * 2);
  __bf16* vT   = (__bf16*)alloc((size_t)N_HEAD * D_HEAD * S_LEN * 2);
  __bf16* qer  = (__bf16*)alloc((size_t)N_HEAD * S_LEN * S_LEN * 2);  // 67 MB (fits L2)
  float*  sa   = (float*)alloc((size_t)S_LEN * D_MOD * 4);            //  4 MB
  float*  x1   = (float*)alloc(RS * D_MOD * 4);                       //  8 MB
  __bf16* x1bf = (__bf16*)alloc(RS * D_MOD * 2);
  __bf16* hbf  = (__bf16*)alloc(RS * D_FF * 2);                       // 16 MB
  float*  ffw  = (float*)alloc(RS * D_MOD * 4);                       //  8 MB
  (void)ws_size; (void)in_sizes; (void)n_in; (void)out_size;

  // --- pack / cast ---
  k_cast_x<<<(unsigned)(RS * D_MOD / 256), 256, 0, stream>>>(src, xbf);
  k_tcast<<<(D_MOD * D_MOD + 255) / 256, 256, 0, stream>>>(Wq, Wqt, D_MOD, D_MOD);
  k_tcast<<<(D_MOD * D_MOD + 255) / 256, 256, 0, stream>>>(Wk, Wkt, D_MOD, D_MOD);
  k_tcast<<<(D_MOD * D_MOD + 255) / 256, 256, 0, stream>>>(Wv, Wvt, D_MOD, D_MOD);
  k_tcast<<<(D_MOD * D_FF + 255) / 256, 256, 0, stream>>>(W1, W1t, D_MOD, D_FF);
  k_tcast<<<(D_FF * D_MOD + 255) / 256, 256, 0, stream>>>(W2, W2t, D_FF, D_MOD);
  k_cast_er<<<(S_LEN * D_HEAD + 255) / 256, 256, 0, stream>>>(Er, erb);

  // --- QKV (batch 0), head-major layouts ---
  k_qkv<<<dim3(D_MOD / 64, S_LEN / 16, 3), 32, 0, stream>>>(
      xbf, Wqt, Wkt, Wvt, bq, bk, bv, qb, kb, vT);

  // --- QEr materialized per head (bf16, L2-resident) ---
  k_qer<<<dim3(S_LEN / 64, S_LEN / 16, N_HEAD), 32, 0, stream>>>(qb, erb, qer);

  // --- flash attention with relative-position skew ---
  k_attn<<<dim3(S_LEN / 16, N_HEAD), 32, 0, stream>>>(qb, kb, vT, qer, sa);

  // --- LN1 (residual + broadcast sa) ---
  k_ln1<<<(unsigned)RS, 32, 0, stream>>>(src, sa, g1, be1, x1, x1bf);

  // --- FFN ---
  k_ff1<<<dim3(D_FF / 64, (unsigned)(RS / 16)), 32, 0, stream>>>(x1bf, W1t, b1, hbf);
  k_ff2<<<dim3(D_MOD / 64, (unsigned)(RS / 16)), 32, 0, stream>>>(hbf, W2t, b2, ffw);

  // --- LN2 + output transpose back to [S,B,D] ---
  k_ln2<<<(unsigned)RS, 32, 0, stream>>>(x1, ffw, g2, be2, out);
}
